// NFFB_30391188586867
// MI455X (gfx1250) — matrix operand order repaired and checked by reading, not compiled, not run on hardware
//
#include <hip/hip_runtime.h>
#include <hip/hip_bf16.h>

typedef __attribute__((ext_vector_type(16))) _Float16 v16h;
typedef __attribute__((ext_vector_type(8)))  float    v8f;

#define NPTS 262144

// workspace layout (in _Float16 units); all regions are per-layer contiguous
//  wsA : frag fi = l*8 + jt*2+kc              -> 64 frags  * 512
//  b0A : frag fi = (l*2+kcl)*8 + mt           -> 128 frags * 512
//  b1A : frag fi = kc*4 + mt                  -> 16 frags  * 512
#define WSA_OFF 0
#define B0A_OFF 32768
#define B1A_OFF 98304
#define TOTAL_HALVES 106496

struct GridPtrs { const float* g[8]; };

// ---------------------------------------------------------------------------
// Prep: f32 -> f16 and pre-swizzle into WMMA A-fragment order.
// A-fragment as packed here (contiguous k; B uses the same order): lane n -> M = n&15, hi = n>>4,
// element e -> K = e + 16*hi.
// ---------------------------------------------------------------------------
__global__ __launch_bounds__(256) void pack_weights_kernel(
    const float* __restrict__ Ws, const float* __restrict__ B0,
    const float* __restrict__ B1, _Float16* __restrict__ out)
{
    int t = blockIdx.x * blockDim.x + threadIdx.x;
    if (t >= TOTAL_HALVES) return;
    float val = 0.f;
    if (t < B0A_OFF) {                       // Ws: A[j][k] = Ws[l][k][j]
        int fi = t >> 9, rr = t & 511, lane = rr >> 4, e = rr & 15;
        int kc = fi & 1, jt = (fi >> 1) & 3, l = fi >> 3;
        int M = lane & 15, hi = lane >> 4;
        int K = e + 16 * hi;
        val = Ws[l * 4096 + (32 * kc + K) * 64 + (16 * jt + M)];
    } else if (t < B1A_OFF) {                // B0: A[m][k] = B0[k][m], layer-major
        int idx = t - B0A_OFF;
        int fi = idx >> 9, rr = idx & 511, lane = rr >> 4, e = rr & 15;
        int mt = fi & 7, kcg = fi >> 3;      // kcg = l*2 + kcl
        int M = lane & 15, hi = lane >> 4;
        int K = e + 16 * hi;
        val = B0[(32 * kcg + K) * 128 + (16 * mt + M)];
    } else {                                 // B1: A[m][k] = B1[k][m]
        int idx = t - B1A_OFF;
        int fi = idx >> 9, rr = idx & 511, lane = rr >> 4, e = rr & 15;
        int mt = fi & 3, kc = fi >> 2;
        int M = lane & 15, hi = lane >> 4;
        int K = e + 16 * hi;
        val = B1[(32 * kc + K) * 64 + (16 * mt + M)];
    }
    out[t] = (_Float16)val;
}

// ---------------------------------------------------------------------------
// Fused NFFB kernel. 256 threads = 8 waves; each wave owns ONE 16-point tile
// (lane n and n+16 duplicate point n&15). Per layer, the block stages the
// layer's weight fragments (24KB) in LDS once; all 8 waves stream them from
// LDS, keeping live VGPRs low. Weights ride in the WMMA A operand,
// activations in B; the D(C-layout) -> next-B conversion is a lane^16
// exchange only.
// ---------------------------------------------------------------------------
__global__ __launch_bounds__(256, 4) void nffb_kernel(
    const float* __restrict__ pos, GridPtrs grids,
    const float* __restrict__ Win, const float* __restrict__ b_in,
    const float* __restrict__ bs,  const float* __restrict__ Amat,
    const float* __restrict__ avec,
    const float* __restrict__ c0,  const float* __restrict__ c1,
    const float* __restrict__ B2,  const float* __restrict__ c2,
    const _Float16* __restrict__ wsp, float* __restrict__ out)
{
    __shared__ _Float16 smem[12288];        // 24 KB staging buffer
    v16h* sv = (v16h*)smem;

    const int tid  = threadIdx.x;
    const int lane = tid & 31;
    const int waveIdx = blockIdx.x * 8 + (tid >> 5);
    const int hi  = lane >> 4;
    const int p16 = lane & 15;

    const v16h* gW  = (const v16h*)(wsp + WSA_OFF);
    const v16h* gB0 = (const v16h*)(wsp + B0A_OFF);
    const v16h* gB1 = (const v16h*)(wsp + B1A_OFF);

    const int p = waveIdx * 16 + p16;
    const float x0 = (pos[p * 2 + 0] - 0.5f) * 2.f;
    const float x1 = (pos[p * 2 + 1] - 0.5f) * 2.f;

    // h0 = sin(30*(x@Win + b)) built directly in B-fragment layout
    v16h hB[2];
#pragma unroll
    for (int kc = 0; kc < 2; ++kc) {
        v16h f;
#pragma unroll
        for (int e = 0; e < 16; ++e) {
            int k = 32 * kc + 16 * hi + e;
            f[e] = (_Float16)__sinf(30.0f * (x0 * Win[k] + x1 * Win[64 + k] + b_in[k]));
        }
        hB[kc] = f;
    }

    // out0 accumulators (128 wide), bias-initialized
    v8f acc0[8];
#pragma unroll
    for (int mt = 0; mt < 8; ++mt)
#pragma unroll
        for (int r = 0; r < 8; ++r)
            acc0[mt][r] = c0[16 * mt + 8 * hi + r];

    float freq = 3.14159265358979323f;      // pi * 2^l
#pragma unroll 1
    for (int l = 0; l < 8; ++l) {
        // ---- stage this layer's fragments in LDS (8 frags Ws + 16 frags B0)
        __syncthreads();                     // previous layer fully consumed
        {
            const v16h* s0 = gW  + l * 256;  // 8 frags * 32
            const v16h* s1 = gB0 + l * 512;  // 16 frags * 32
            sv[tid]       = s0[tid];
            sv[256 + tid] = s1[tid];
            sv[512 + tid] = s1[256 + tid];
        }
        __syncthreads();

        // ---- d = bias; d += Ws^T x h   (D rows = neurons, cols = points)
        v8f d[4];
#pragma unroll
        for (int jt = 0; jt < 4; ++jt)
#pragma unroll
            for (int r = 0; r < 8; ++r)
                d[jt][r] = bs[l * 64 + 16 * jt + 8 * hi + r];
#pragma unroll
        for (int jt = 0; jt < 4; ++jt)
#pragma unroll
            for (int kc = 0; kc < 2; ++kc) {
                v16h wA = sv[(jt * 2 + kc) * 32 + lane];
                d[jt] = __builtin_amdgcn_wmma_f32_16x16x32_f16(
                    false, wA, false, hB[kc], (short)0, d[jt], false, false);
            }

        float s[4][8];
#pragma unroll
        for (int jt = 0; jt < 4; ++jt)
#pragma unroll
            for (int r = 0; r < 8; ++r) s[jt][r] = __sinf(d[jt][r]);

        // ---- bilerp grid l at this lane's point
        const float* gl = grids.g[l];
        const int R = 16 << l;
        float gf0, gf1;
        {
            float Rm1 = (float)(R - 1);
            float u = fminf(fmaxf((x0 + 1.f) * 0.5f * Rm1, 0.f), Rm1);
            float v = fminf(fmaxf((x1 + 1.f) * 0.5f * Rm1, 0.f), Rm1);
            int ix0 = (int)u, iy0 = (int)v;
            int ix1 = min(ix0 + 1, R - 1), iy1 = min(iy0 + 1, R - 1);
            float fx = u - (float)ix0, fy = v - (float)iy0;
            const float* p00 = gl + (ix0 * R + iy0) * 2;
            const float* p01 = gl + (ix0 * R + iy1) * 2;
            const float* p10 = gl + (ix1 * R + iy0) * 2;
            const float* p11 = gl + (ix1 * R + iy1) * 2;
            float w00 = (1.f - fx) * (1.f - fy), w01 = (1.f - fx) * fy;
            float w10 = fx * (1.f - fy),         w11 = fx * fy;
            gf0 = p00[0] * w00 + p01[0] * w01 + p10[0] * w10 + p11[0] * w11;
            gf1 = p00[1] * w00 + p01[1] * w01 + p10[1] * w10 + p11[1] * w11;
        }

        // ---- D -> B via half-wave exchange; feats = h * filt
        const float* A0 = Amat + (l * 2 + 0) * 64;
        const float* A1 = Amat + (l * 2 + 1) * 64;
        const float* av = avec + l * 64;
        v16h featB[2];
#pragma unroll
        for (int kc = 0; kc < 2; ++kc) {
            v16h hb, fb;
#pragma unroll
            for (int r = 0; r < 8; ++r) {
                float send = hi ? s[2 * kc][r] : s[2 * kc + 1][r];
                float recv = __shfl_xor(send, 16, 32);
                float e_lo = hi ? recv : s[2 * kc][r];
                float e_hi = hi ? s[2 * kc + 1][r] : recv;
                int k_lo = 32 * kc + 16 * hi + r;
                int k_hi = k_lo + 8;
                float f_lo = __sinf(freq * (gf0 * A0[k_lo] + gf1 * A1[k_lo] + av[k_lo]));
                float f_hi = __sinf(freq * (gf0 * A0[k_hi] + gf1 * A1[k_hi] + av[k_hi]));
                hb[r]     = (_Float16)e_lo;
                hb[8 + r] = (_Float16)e_hi;
                fb[r]     = (_Float16)(e_lo * f_lo);
                fb[8 + r] = (_Float16)(e_hi * f_hi);
            }
            hB[kc]    = hb;
            featB[kc] = fb;
        }

        // ---- stream this layer's B0 slice from LDS, accumulate into out0
#pragma unroll
        for (int kcl = 0; kcl < 2; ++kcl)
#pragma unroll
            for (int mt = 0; mt < 8; ++mt) {
                v16h bA = sv[256 + (kcl * 8 + mt) * 32 + lane];
                acc0[mt] = __builtin_amdgcn_wmma_f32_16x16x32_f16(
                    false, bA, false, featB[kcl], (short)0, acc0[mt], false, false);
            }
        freq *= 2.0f;
    }

    // ---- stage B1 fragments (16 KB), then out1 = relu(out0) @ B1 + c1
    __syncthreads();
    sv[tid]       = gB1[tid];
    sv[256 + tid] = gB1[256 + tid];
    __syncthreads();

    v8f acc1[4];
#pragma unroll
    for (int mt = 0; mt < 4; ++mt)
#pragma unroll
        for (int r = 0; r < 8; ++r)
            acc1[mt][r] = c1[16 * mt + 8 * hi + r];

    v16h o0B[4];
#pragma unroll
    for (int kc = 0; kc < 4; ++kc) {
        v16h f;
#pragma unroll
        for (int r = 0; r < 8; ++r) {
            float slo = fmaxf(acc0[2 * kc][r], 0.f);
            float shi = fmaxf(acc0[2 * kc + 1][r], 0.f);
            float send = hi ? slo : shi;
            float recv = __shfl_xor(send, 16, 32);
            f[r]     = (_Float16)(hi ? recv : slo);
            f[8 + r] = (_Float16)(hi ? shi : recv);
        }
        o0B[kc] = f;
    }
#pragma unroll
    for (int kc = 0; kc < 4; ++kc)
#pragma unroll
        for (int mt = 0; mt < 4; ++mt) {
            v16h bA = sv[(kc * 4 + mt) * 32 + lane];
            acc1[mt] = __builtin_amdgcn_wmma_f32_16x16x32_f16(
                false, bA, false, o0B[kc], (short)0, acc1[mt], false, false);
        }

    // ---- out = clip(relu(out1) @ B2 + c2)  (64 -> 3 VALU dot + half reduce)
    float o[3] = {0.f, 0.f, 0.f};
#pragma unroll
    for (int mt = 0; mt < 4; ++mt)
#pragma unroll
        for (int r = 0; r < 8; ++r) {
            float v = fmaxf(acc1[mt][r], 0.f);
            int m = 16 * mt + 8 * hi + r;
            o[0] += v * B2[m * 3 + 0];
            o[1] += v * B2[m * 3 + 1];
            o[2] += v * B2[m * 3 + 2];
        }
#pragma unroll
    for (int j = 0; j < 3; ++j) {
        o[j] += __shfl_xor(o[j], 16, 32);
        o[j] = fminf(fmaxf(o[j] + c2[j], -1.f), 1.f);
    }
    if (hi == 0) {
        out[p * 3 + 0] = o[0];
        out[p * 3 + 1] = o[1];
        out[p * 3 + 2] = o[2];
    }
}

extern "C" void kernel_launch(void* const* d_in, const int* in_sizes, int n_in,
                              void* d_out, int out_size, void* d_ws, size_t ws_size,
                              hipStream_t stream) {
    const float* pos  = (const float*)d_in[0];
    GridPtrs grids;
    for (int l = 0; l < 8; ++l) grids.g[l] = (const float*)d_in[1 + l];
    const float* Win  = (const float*)d_in[9];
    const float* b_in = (const float*)d_in[10];
    const float* Ws   = (const float*)d_in[11];
    const float* bs   = (const float*)d_in[12];
    const float* Amat = (const float*)d_in[13];
    const float* avec = (const float*)d_in[14];
    const float* B0   = (const float*)d_in[15];
    const float* c0   = (const float*)d_in[16];
    const float* B1   = (const float*)d_in[17];
    const float* c1   = (const float*)d_in[18];
    const float* B2   = (const float*)d_in[19];
    const float* c2   = (const float*)d_in[20];
    float* outp       = (float*)d_out;
    _Float16* wsp     = (_Float16*)d_ws;

    pack_weights_kernel<<<(TOTAL_HALVES + 255) / 256, 256, 0, stream>>>(Ws, B0, B1, wsp);
    nffb_kernel<<<NPTS / 128, 256, 0, stream>>>(pos, grids, Win, b_in, bs, Amat, avec,
                                                c0, c1, B2, c2, wsp, outp);
}
